// TemporalAttentionDecoder_prev_decoder_states_26680336843494
// MI455X (gfx1250) — matrix operand order repaired and checked by reading, not compiled, not run on hardware
//
#include <hip/hip_runtime.h>
#include <hip/hip_bf16.h>

// MI455X / gfx1250, wave32. Sizes fixed by the reference.
#define B_ 4096
#define L_ 64
#define M_ 128
#define P_ 128
#define T_ 64

typedef __attribute__((ext_vector_type(16))) _Float16 v16h;
typedef __attribute__((ext_vector_type(8)))  float    v8f;
typedef __attribute__((ext_vector_type(8)))  _Float16 h8;
typedef __attribute__((ext_vector_type(4)))  float    f4;   // native clang vec
                                                            // (HIP float4 is a
                                                            // class; NT builtin
                                                            // needs a real vec)

__device__ __forceinline__ float sigmoidf_(float x) { return 1.0f / (1.0f + __expf(-x)); }

// Overflow-safe fast tanh: one v_exp + a few VALU ops instead of libm tanhf.
// Keeps the HBM-bound attention loop from becoming TRANS/VALU-bound.
__device__ __forceinline__ float fast_tanh(float x) {
  float ax = fabsf(x);
  float e = __expf(-2.0f * ax);            // (0, 1]
  float r = 1.0f - 2.0f * e / (1.0f + e);  // tanh(|x|)
  return copysignf(r, x);
}

// ---- WMMA fragment helpers (contiguous k order, same in A and B, wave32) ----
// A (16x32 f16) from LDS: lanes 0-15 rows 0-15 K=k0..k0+15 (lanes 16-31 same
// rows, K=k0+16..k0+31).
__device__ __forceinline__ v16h a_frag_lds(const _Float16* sh, int ld, int k0) {
  const int lane = threadIdx.x & 31;
  const int m = lane & 15;
  const int hi = (lane >> 4) & 1;
  const _Float16* row = sh + m * ld + k0 + hi * 16;
  v16h a;
#pragma unroll
  for (int i = 0; i < 8; ++i) a[i] = row[i];
#pragma unroll
  for (int i = 0; i < 8; ++i) a[8 + i] = row[8 + i];
  return a;
}

// B (32x16 f16) from PRE-CONVERTED f16 weights in workspace: lane holds
// column n; lanes 0-15 K=k0..k0+15, lanes 16-31 K=k0+16..k0+31.  Row-major
// [n_out][k] weights give each lane 16 contiguous f16 = two aligned 16-byte
// vector loads, no f32->f16 converts in the inner loop.
__device__ __forceinline__ v16h b_frag_w16(const _Float16* __restrict__ W,
                                           long ldk, int n, int k0) {
  const int lane = threadIdx.x & 31;
  const int hi = (lane >> 4) & 1;
  const _Float16* col = W + (long)n * ldk + k0 + hi * 16;
  h8 lo = *(const h8*)(col);
  h8 hv = *(const h8*)(col + 8);
  v16h b;
#pragma unroll
  for (int i = 0; i < 8; ++i) { b[i] = lo[i]; b[8 + i] = hv[i]; }
  return b;
}

__device__ __forceinline__ v8f wmma_(v16h a, v16h b, v8f c) {
  return __builtin_amdgcn_wmma_f32_16x16x32_f16(false, a, false, b, (short)0, c,
                                                false, false);
}

// ============================================================================
// Kernel 0: one-time weight f32 -> f16 conversion (loop-invariant operands).
// ============================================================================
__global__ __launch_bounds__(256) void cvt_kernel(const float* __restrict__ src,
                                                  _Float16* __restrict__ dst,
                                                  int n) {
  int i = blockIdx.x * 256 + threadIdx.x;
  if (i < n) dst[i] = (_Float16)src[i];
}

// ============================================================================
// Kernel 1: y1[b,l,n] = sum_m enc[b,l,m] * U[n,m]   (stored f16: 67 MB stream
// read every step; small enough to stay L2-resident if enc reads are NT).
// ============================================================================
__global__ __launch_bounds__(256) void y1_kernel(const float* __restrict__ enc,
                                                 const _Float16* __restrict__ Uh,
                                                 _Float16* __restrict__ y1h) {
  __shared__ _Float16 shA[16][128];
  const int tid = threadIdx.x;
  const long r0 = (long)blockIdx.x * 16;
#pragma unroll
  for (int i = 0; i < 8; ++i) {
    int idx = tid + 256 * i;  // 2048 elements
    int r = idx >> 7, c = idx & 127;
    shA[r][c] = (_Float16)enc[(r0 + r) * 128 + c];
  }
  __syncthreads();
  const int wid = tid >> 5, lane = tid & 31;
  const int hi = (lane >> 4) & 1;
  const int n = wid * 16 + (lane & 15);
  v8f acc = {};
#pragma unroll
  for (int k0 = 0; k0 < 128; k0 += 32) {
    v16h a = a_frag_lds(&shA[0][0], 128, k0);
    v16h b = b_frag_w16(Uh, 128, n, k0);
    acc = wmma_(a, b, acc);
  }
#pragma unroll
  for (int r = 0; r < 8; ++r)
    y1h[(r0 + r + hi * 8) * 128 + n] = (_Float16)acc[r];
}

// ============================================================================
// Kernel 2: one full scan step, fused.  Block = 16 batch rows, 256 thr, 8 waves.
// ============================================================================
__global__ __launch_bounds__(256) void step_kernel(
    const float* __restrict__ enc, const float* __restrict__ y,
    const _Float16* __restrict__ Wdh, const float* __restrict__ W_d_b,
    const float* __restrict__ v_d_w, const float* __restrict__ w_tilda_w,
    const float* __restrict__ w_tilda_b, const float* __restrict__ W_ih,
    const float* __restrict__ b_ih, const _Float16* __restrict__ Whhh,
    const float* __restrict__ b_hh, const _Float16* __restrict__ Wdech,
    const float* __restrict__ W_dec_b, const _Float16* __restrict__ y1h,
    float* __restrict__ dbuf, float* __restrict__ sbuf,
    float* __restrict__ dfin, float* __restrict__ clast, int t) {
  __shared__ _Float16 shA[16][256];   // [d|s] f16 (WMMA A)
  __shared__ float    shS[16][128];   // s f32 for LSTM update
  __shared__ float    shX[16][128];   // x1
  __shared__ float    shLB[16][64];   // logits -> beta
  __shared__ float    shC[16][128];   // c_t
  __shared__ _Float16 shDn[16][128];  // d_new f16 (WMMA A)
  __shared__ float    shV[128];
  __shared__ float    shWt[129];
  __shared__ float    shYt[16];

  const int tid = threadIdx.x;
  const int wid = tid >> 5, lane = tid & 31;
  const int hi = (lane >> 4) & 1;
  const long b0 = (long)blockIdx.x * 16;

  // -- p0: stage previous state (zeros at t==0 -> never reads poisoned ws) --
#pragma unroll
  for (int i = 0; i < 8; ++i) {
    int idx = tid + 256 * i;
    int r = idx >> 7, c = idx & 127;
    float dv = (t == 0) ? 0.0f : dbuf[(b0 + r) * 128 + c];
    float sv = (t == 0) ? 0.0f : sbuf[(b0 + r) * 128 + c];
    shA[r][c] = (_Float16)dv;
    shA[r][128 + c] = (_Float16)sv;
    shS[r][c] = sv;
  }
  if (tid < 128) shV[tid] = v_d_w[tid];
  if (tid < 129) shWt[tid] = w_tilda_w[tid];
  __syncthreads();

  // -- p1: x1 = [d|s] @ W_d.T + b_d  (WMMA, K=256) --
  {
    const int n = wid * 16 + (lane & 15);
    v8f acc = {};
#pragma unroll
    for (int k0 = 0; k0 < 256; k0 += 32) {
      v16h a = a_frag_lds(&shA[0][0], 256, k0);
      v16h b = b_frag_w16(Wdh, 256, n, k0);
      acc = wmma_(a, b, acc);
    }
    float bias = W_d_b[n];
#pragma unroll
    for (int r = 0; r < 8; ++r) shX[r + hi * 8][n] = acc[r] + bias;
  }
  __syncthreads();

  // -- p2: logits[b,l] = sum_m v[m] * tanh(x1[b,m] + y1[b,l,m]) --
  // Dominant HBM/L2 stream; fast_tanh keeps it memory-bound.
#pragma unroll
  for (int i = 0; i < 4; ++i) {
    int pair = tid + 256 * i;  // 16*64 = 1024 (b,l) pairs
    int bl = pair >> 6, l = pair & 63;
    const _Float16* yr = y1h + ((b0 + bl) * 64 + l) * 128;
    const float* xr = shX[bl];
    float acc = 0.0f;
    for (int m = 0; m < 128; m += 8) {
      h8 yv = *(const h8*)(yr + m);
#pragma unroll
      for (int j = 0; j < 8; ++j)
        acc += shV[m + j] * fast_tanh(xr[m + j] + (float)yv[j]);
    }
    shLB[bl][l] = acc;
  }
  __syncthreads();

  // -- p3: softmax over L=64, one lane per batch row --
  if (tid < 16) {
    float mx = -1e30f;
    for (int l = 0; l < 64; ++l) mx = fmaxf(mx, shLB[tid][l]);
    float sum = 0.0f;
    for (int l = 0; l < 64; ++l) {
      float e = __expf(shLB[tid][l] - mx);
      shLB[tid][l] = e;
      sum += e;
    }
    float inv = 1.0f / sum;
    for (int l = 0; l < 64; ++l) shLB[tid][l] *= inv;
  }
  __syncthreads();

  // -- p4: c_t[b,m] = sum_l beta[b,l] * enc[b,l,m] --
  // enc is streamed once per step and not reused within the step: load NT so
  // the 134 MB/step stream does not evict the L2-resident y1.
#pragma unroll
  for (int i = 0; i < 2; ++i) {
    int task = tid + 256 * i;  // 16 rows * 32 float4 groups
    int bl = task >> 5, m0 = (task & 31) * 4;
    f4 acc = {0.f, 0.f, 0.f, 0.f};
    const f4* er = (const f4*)(enc + ((b0 + bl) * 64) * 128 + m0);
    for (int l = 0; l < 64; ++l) {
      float be = shLB[bl][l];
      f4 e = __builtin_nontemporal_load(er + l * 32);  // 32 f4 per row
      acc += be * e;
    }
    shC[bl][m0] = acc.x; shC[bl][m0 + 1] = acc.y;
    shC[bl][m0 + 2] = acc.z; shC[bl][m0 + 3] = acc.w;
    *(f4*)(clast + (b0 + bl) * 128 + m0) = acc;  // last step's value survives
  }
  __syncthreads();

  // -- p4b: y_til[b] (scalar per row) --
  if (tid < 16) {
    float acc = 0.0f;
    for (int m = 0; m < 128; ++m) acc += shC[tid][m] * shWt[m];
    acc += shWt[128] * y[(b0 + tid) * T_ + t] + w_tilda_b[0];
    shYt[tid] = acc;
  }
  __syncthreads();

  // -- p5+p6: gates = d @ W_hh.T (WMMA); wave owns same 16 cols of i/f/g/o,
  //           so the LSTM pointwise consumes accumulators in registers. --
  {
    const int j = wid * 16 + (lane & 15);  // gate column 0..127
    v8f aI = {}, aF = {}, aG = {}, aO = {};
#pragma unroll
    for (int k0 = 0; k0 < 128; k0 += 32) {
      v16h a = a_frag_lds(&shA[0][0], 256, k0);  // d part of [d|s]
      aI = wmma_(a, b_frag_w16(Whhh, 128, j, k0), aI);
      aF = wmma_(a, b_frag_w16(Whhh, 128, j + 128, k0), aF);
      aG = wmma_(a, b_frag_w16(Whhh, 128, j + 256, k0), aG);
      aO = wmma_(a, b_frag_w16(Whhh, 128, j + 384, k0), aO);
    }
    float wIi = W_ih[j], wIf = W_ih[128 + j], wIg = W_ih[256 + j], wIo = W_ih[384 + j];
    float bI = b_ih[j] + b_hh[j];
    float bF = b_ih[128 + j] + b_hh[128 + j];
    float bG = b_ih[256 + j] + b_hh[256 + j];
    float bO = b_ih[384 + j] + b_hh[384 + j];
#pragma unroll
    for (int r = 0; r < 8; ++r) {
      int m = r + hi * 8;
      float yti = shYt[m];
      float ig = aI[r] + yti * wIi + bI;
      float fg = aF[r] + yti * wIf + bF;
      float gg = aG[r] + yti * wIg + bG;
      float og = aO[r] + yti * wIo + bO;
      float sv = shS[m][j];
      float sn = sigmoidf_(fg) * sv + sigmoidf_(ig) * fast_tanh(gg);
      float dn = sigmoidf_(og) * fast_tanh(sn);
      dbuf[(b0 + m) * 128 + j] = dn;
      sbuf[(b0 + m) * 128 + j] = sn;
      shDn[m][j] = (_Float16)dn;
    }
  }
  __syncthreads();

  // -- p7: d_fin += d_new @ W_dec[:, P*(t+1):P*(t+2)].T  (WMMA, running acc) --
  {
    const int n = wid * 16 + (lane & 15);
    v8f acc;
#pragma unroll
    for (int r = 0; r < 8; ++r)
      acc[r] = (t == 0) ? W_dec_b[n] : dfin[(b0 + r + hi * 8) * 128 + n];
    const _Float16* Wdec = Wdech + (long)128 * (t + 1);  // ldk = P*(T+1) = 8320
#pragma unroll
    for (int k0 = 0; k0 < 128; k0 += 32) {
      v16h a = a_frag_lds(&shDn[0][0], 128, k0);
      v16h b = b_frag_w16(Wdec, 8320, n, k0);
      acc = wmma_(a, b, acc);
    }
#pragma unroll
    for (int r = 0; r < 8; ++r)
      dfin[(b0 + r + hi * 8) * 128 + n] = acc[r];
  }
}

// ============================================================================
// Kernel 3: h = [d_fin|c_last] @ W_y.T + b ; out = h @ v_y.T + b
// ============================================================================
__global__ __launch_bounds__(256) void final_kernel(
    const float* __restrict__ dfin, const float* __restrict__ clast,
    const _Float16* __restrict__ Wyh, const float* __restrict__ W_y_b,
    const float* __restrict__ v_y_w, const float* __restrict__ v_y_b,
    float* __restrict__ out) {
  __shared__ _Float16 shA[16][256];
  __shared__ float shH[16][128];
  const int tid = threadIdx.x;
  const int wid = tid >> 5, lane = tid & 31;
  const int hi = (lane >> 4) & 1;
  const long b0 = (long)blockIdx.x * 16;
#pragma unroll
  for (int i = 0; i < 8; ++i) {
    int idx = tid + 256 * i;
    int r = idx >> 7, c = idx & 127;
    shA[r][c] = (_Float16)dfin[(b0 + r) * 128 + c];
    shA[r][128 + c] = (_Float16)clast[(b0 + r) * 128 + c];
  }
  __syncthreads();
  {
    const int n = wid * 16 + (lane & 15);
    v8f acc = {};
#pragma unroll
    for (int k0 = 0; k0 < 256; k0 += 32) {
      v16h a = a_frag_lds(&shA[0][0], 256, k0);
      v16h b = b_frag_w16(Wyh, 256, n, k0);
      acc = wmma_(a, b, acc);
    }
    float bias = W_y_b[n];
#pragma unroll
    for (int r = 0; r < 8; ++r) shH[r + hi * 8][n] = acc[r] + bias;
  }
  __syncthreads();
  if (tid < 16) {
    float acc = 0.0f;
    for (int m = 0; m < 128; ++m) acc += shH[tid][m] * v_y_w[m];
    out[b0 + tid] = acc + v_y_b[0];
  }
}

// ============================================================================
extern "C" void kernel_launch(void* const* d_in, const int* in_sizes, int n_in,
                              void* d_out, int out_size, void* d_ws, size_t ws_size,
                              hipStream_t stream) {
  (void)in_sizes; (void)n_in; (void)out_size; (void)ws_size;
  const float* enc       = (const float*)d_in[0];
  const float* y         = (const float*)d_in[1];
  const float* W_d_w     = (const float*)d_in[2];
  const float* W_d_b     = (const float*)d_in[3];
  const float* U_d_w     = (const float*)d_in[4];
  const float* v_d_w     = (const float*)d_in[5];
  const float* w_tilda_w = (const float*)d_in[6];
  const float* w_tilda_b = (const float*)d_in[7];
  const float* W_ih      = (const float*)d_in[8];
  const float* b_ih      = (const float*)d_in[9];
  const float* W_hh      = (const float*)d_in[10];
  const float* b_hh      = (const float*)d_in[11];
  const float* W_dec_w   = (const float*)d_in[12];
  const float* W_dec_b   = (const float*)d_in[13];
  const float* W_y_w     = (const float*)d_in[14];
  const float* W_y_b     = (const float*)d_in[15];
  const float* v_y_w     = (const float*)d_in[16];
  const float* v_y_b     = (const float*)d_in[17];
  float* out = (float*)d_out;

  // Workspace layout (~78 MB): y1 f16 stream, f32 state, f16 weight copies.
  char* ws = (char*)d_ws;
  _Float16* y1h = (_Float16*)ws;                              // B*L*M f16 = 64 MiB
  float* dbuf  = (float*)(ws + (size_t)B_ * L_ * M_ * 2);     // B*P
  float* sbuf  = dbuf + (size_t)B_ * P_;                      // B*P
  float* dfin  = sbuf + (size_t)B_ * P_;                      // B*P
  float* clast = dfin + (size_t)B_ * P_;                      // B*M
  _Float16* Uh    = (_Float16*)(clast + (size_t)B_ * M_);     // 128*128
  _Float16* Wdh   = Uh + 128 * 128;                           // 128*256
  _Float16* Whhh  = Wdh + 128 * 256;                          // 512*128
  _Float16* Wdech = Whhh + 512 * 128;                         // 128*8320
  _Float16* Wyh   = Wdech + (size_t)128 * 8320;               // 128*256

  // One-time f32 -> f16 weight conversion (removes per-step VALU converts).
  cvt_kernel<<<(128 * 128 + 255) / 256, 256, 0, stream>>>(U_d_w, Uh, 128 * 128);
  cvt_kernel<<<(128 * 256 + 255) / 256, 256, 0, stream>>>(W_d_w, Wdh, 128 * 256);
  cvt_kernel<<<(512 * 128 + 255) / 256, 256, 0, stream>>>(W_hh, Whhh, 512 * 128);
  cvt_kernel<<<(128 * 8320 + 255) / 256, 256, 0, stream>>>(W_dec_w, Wdech, 128 * 8320);
  cvt_kernel<<<(128 * 256 + 255) / 256, 256, 0, stream>>>(W_y_w, Wyh, 128 * 256);

  y1_kernel<<<(B_ * L_) / 16, 256, 0, stream>>>(enc, Uh, y1h);
  for (int t = 0; t < T_; ++t) {
    step_kernel<<<B_ / 16, 256, 0, stream>>>(
        enc, y, Wdh, W_d_b, v_d_w, w_tilda_w, w_tilda_b, W_ih, b_ih, Whhh,
        b_hh, Wdech, W_dec_b, y1h, dbuf, sbuf, dfin, clast, t);
  }
  final_kernel<<<B_ / 16, 256, 0, stream>>>(dfin, clast, Wyh, W_y_b, v_y_w,
                                            v_y_b, out);
}
